// IsInOneNetworkTrainer_46402826666699
// MI455X (gfx1250) — matrix operand order repaired and checked by reading, not compiled, not run on hardware
//
#include <hip/hip_runtime.h>
#include <hip/hip_bf16.h>

typedef __attribute__((ext_vector_type(16))) __bf16 v16bf;
typedef __attribute__((ext_vector_type(8)))  float  v8f;
typedef __attribute__((ext_vector_type(4)))  float  f32x4;

#define WIRE_DIM 32
#define NUM_WIRES 64
#define FEAT (NUM_WIRES * WIRE_DIM)   // 2048 floats per batch row
#define HID 10

__global__ __launch_bounds__(256) void isin_wmma_kernel(
    const float* __restrict__ outputs,   // (B, 2048) f32
    const int*   __restrict__ tests,     // (B, 2) int32
    const float* __restrict__ W1,        // (64, 10)
    const float* __restrict__ b1,        // (10,)
    const float* __restrict__ W2,        // (10,)  (W2[:,0])
    float*       __restrict__ out,       // (B,)
    int nbatch)
{
    const int lane = threadIdx.x & 31;
    const int wave = threadIdx.x >> 5;
    const int b    = blockIdx.x * 8 + wave;
    if (b >= nbatch) return;             // wave-uniform: EXEC stays all-ones for WMMA

    const int row  = lane & 15;          // A row / D column / B column index
    const int half = lane >> 4;          // lane-half select
    const int kb   = half * 8;           // A-matrix K base (ISA 16-bit A 16x32 layout)
    const int kh   = half * 16;          // unused: B loads use kb (lanes 16-31 hold K=8..15, 24..31)

    // ---- B matrices: W1 lower/upper 32 rows in WMMA-B layout ----
    // Columns n>=10 load a clamped replica of column 9 instead of zeros: their
    // (finite) garbage in D is killed by w2v==0 when forming logits, so no
    // per-element zero-select is needed.
    const int n     = row;
    const int nload = n < HID ? n : (HID - 1);   // in-bounds clamped column
    v16bf B1v, B2v;
    #pragma unroll
    for (int e = 0; e < 16; ++e) {
        const int K = kb + (e & 7) + ((e >> 3) << 4);
        B1v[e] = (__bf16)W1[K * HID + nload];
        B2v[e] = (__bf16)W1[(WIRE_DIM + K) * HID + nload];
    }
    const float w2l = W2[nload];
    const float w2v = (n < HID) ? w2l : 0.0f;    // the only padding select we need
    const float b1v = b1[nload];                 // garbage cols killed by w2v anyway

    const int person   = tests[2 * b];
    const int location = tests[2 * b + 1];
    const float* base  = outputs + (size_t)b * FEAT;

    // ---- A1 = person_vec broadcast to all 16 rows (rows identical -> same per-lane K data) ----
    const float* pv = base + person * WIRE_DIM + kb;
    f32x4 p0 = *(const f32x4*)(pv);
    f32x4 p1 = *(const f32x4*)(pv + 4);
    f32x4 p2 = *(const f32x4*)(pv + 16);
    f32x4 p3 = *(const f32x4*)(pv + 20);
    v16bf A1;
    #pragma unroll
    for (int e = 0; e < 4; ++e) {
        A1[e]      = (__bf16)p0[e];
        A1[4 + e]  = (__bf16)p1[e];
        A1[8 + e]  = (__bf16)p2[e];
        A1[12 + e] = (__bf16)p3[e];
    }

    // C0 = b1 broadcast down columns; Cp = person_vec @ W1[:32] + b1 (shared by all wires)
    v8f C0;
    #pragma unroll
    for (int r = 0; r < 8; ++r) C0[r] = b1v;
    v8f Cp = __builtin_amdgcn_wmma_f32_16x16x32_bf16(
        false, A1, false, B1v, (short)0, C0, false, false);

    // ---- 4 tiles of 16 wires: D = wires @ W1[32:] + Cp ----
    // Packed butterfly row-reduce: each xor stage folds 2 rows into complementary
    // lane groups. Final z: lane bits [3:1] = row bits [0:2] (bit-reversed),
    // lane bit 0 is a duplicate; half selects rows 0-7 vs 8-15 of the tile.
    float zt[4];
    #pragma unroll
    for (int t = 0; t < 4; ++t) {
        const float* wp = base + (t * 16 + row) * WIRE_DIM + kb;
        f32x4 x0 = *(const f32x4*)(wp);
        f32x4 x1 = *(const f32x4*)(wp + 4);
        f32x4 x2 = *(const f32x4*)(wp + 16);
        f32x4 x3 = *(const f32x4*)(wp + 20);
        v16bf A2;
        #pragma unroll
        for (int e = 0; e < 4; ++e) {
            A2[e]      = (__bf16)x0[e];
            A2[4 + e]  = (__bf16)x1[e];
            A2[8 + e]  = (__bf16)x2[e];
            A2[12 + e] = (__bf16)x3[e];
        }
        v8f D = __builtin_amdgcn_wmma_f32_16x16x32_bf16(
            false, A2, false, B2v, (short)0, Cp, false, false);

        float p[8];
        #pragma unroll
        for (int r = 0; r < 8; ++r) {                 // relu * W2, partial over {l, l^8}
            p[r] = fmaxf(D[r], 0.0f) * w2v;
            p[r] += __shfl_xor(p[r], 8, 32);
        }
        float u[4];
        #pragma unroll
        for (int j = 0; j < 4; ++j) {                 // fold row bit0 into lane bit3
            u[j] = (lane & 8) ? p[2 * j + 1] : p[2 * j];
            u[j] += __shfl_xor(u[j], 4, 32);
        }
        float w2r[2];
        #pragma unroll
        for (int j = 0; j < 2; ++j) {                 // fold row bit1 into lane bit2
            w2r[j] = (lane & 4) ? u[2 * j + 1] : u[2 * j];
            w2r[j] += __shfl_xor(w2r[j], 2, 32);
        }
        float z = (lane & 2) ? w2r[1] : w2r[0];       // fold row bit2 into lane bit1
        z += __shfl_xor(z, 1, 32);
        zt[t] = z;                                    // logit of wire t*16 + 8*half + R(lane)
    }

    // ---- log-softmax over 64 distributed logits (4 per lane, lane^1 duplicated) ----
    // Skip xor-1 stage: lanes with equal bit0 cover every distinct wire exactly once.
    float m = fmaxf(fmaxf(zt[0], zt[1]), fmaxf(zt[2], zt[3]));
    m = fmaxf(m, __shfl_xor(m, 2, 32));
    m = fmaxf(m, __shfl_xor(m, 4, 32));
    m = fmaxf(m, __shfl_xor(m, 8, 32));
    m = fmaxf(m, __shfl_xor(m, 16, 32));
    float s = __expf(zt[0] - m) + __expf(zt[1] - m)
            + __expf(zt[2] - m) + __expf(zt[3] - m);
    s += __shfl_xor(s, 2, 32);
    s += __shfl_xor(s, 4, 32);
    s += __shfl_xor(s, 8, 32);
    s += __shfl_xor(s, 16, 32);
    const float lse = m + __logf(s);                  // b2 omitted: cancels in log_softmax

    // ---- fetch logit at `location`: owning lane = bit-reversed row bits ----
    const int lt  = location >> 4;                    // tile
    const int r16 = location & 15;
    const int hn  = r16 >> 3;                         // half
    const int r   = r16 & 7;                          // row within half
    const int src = ((r & 1) << 3) | (((r >> 1) & 1) << 2) | (((r >> 2) & 1) << 1) | (hn << 4);
    float sel = zt[0];
    sel = (lt == 1) ? zt[1] : sel;
    sel = (lt == 2) ? zt[2] : sel;
    sel = (lt == 3) ? zt[3] : sel;
    const float locv = __shfl(sel, src, 32);

    if (lane == 0) out[b] = lse - locv;               // loss = logsumexp - logit[location]
}

extern "C" void kernel_launch(void* const* d_in, const int* in_sizes, int n_in,
                              void* d_out, int out_size, void* d_ws, size_t ws_size,
                              hipStream_t stream) {
    const float* outputs = (const float*)d_in[0];
    const int*   tests   = (const int*)  d_in[1];
    const float* W1      = (const float*)d_in[2];
    const float* b1      = (const float*)d_in[3];
    const float* W2      = (const float*)d_in[4];
    // d_in[5] = b2: uniform shift of all logits -> cancels in log_softmax, unused.
    float* out = (float*)d_out;

    const int nbatch = out_size;                 // one loss per batch row
    const int blocks = (nbatch + 7) / 8;         // 8 waves (batches) per 256-thread block
    isin_wmma_kernel<<<blocks, 256, 0, stream>>>(outputs, tests, W1, b1, W2, out, nbatch);
}
